// StudentGrafomerModel_4758823764605
// MI455X (gfx1250) — matrix operand order repaired and checked by reading, not compiled, not run on hardware
//
#include <hip/hip_runtime.h>
#include <hip/hip_bf16.h>

// ---------------------------------------------------------------------------
// Types & helpers
// ---------------------------------------------------------------------------
typedef __bf16 bf16_t;
typedef __attribute__((ext_vector_type(16))) __bf16 v16bf;
typedef __attribute__((ext_vector_type(8)))  __bf16 v8bf;
typedef __attribute__((ext_vector_type(8)))  float  v8f;
typedef __attribute__((ext_vector_type(4)))  int    i32x4;

typedef i32x4 __attribute__((address_space(1)))* gptr4;   // global int4*
typedef i32x4 __attribute__((address_space(3)))* lptr4;   // LDS int4*

#define DEV static __device__ __forceinline__

#if __has_builtin(__builtin_amdgcn_global_load_async_to_lds_b128)
#define HAS_ASYNC_LDS 1
#endif

DEV bf16_t f2bf(float f) {
  union { float f; unsigned u; } x; x.f = f;
  unsigned r = x.u + 0x7FFFu + ((x.u >> 16) & 1u);   // round-to-nearest-even
  unsigned short h = (unsigned short)(r >> 16);
  union { unsigned short s; bf16_t b; } y; y.s = h;
  return y.b;
}

// 16-byte global->LDS copy: async DMA path on gfx1250, plain copy otherwise.
DEV void copy16_g2l(bf16_t* dst, const bf16_t* src) {
#ifdef HAS_ASYNC_LDS
  __builtin_amdgcn_global_load_async_to_lds_b128((gptr4)(src), (lptr4)(dst), 0, 0);
#else
  *(uint4*)dst = *(const uint4*)src;
#endif
}

DEV void wait_async0() {
#ifdef HAS_ASYNC_LDS
#if __has_builtin(__builtin_amdgcn_s_wait_asynccnt)
  __builtin_amdgcn_s_wait_asynccnt(0);
#else
  asm volatile("s_wait_asynccnt 0" ::: "memory");
#endif
#endif
}

DEV v8f wmma_bf16(v16bf a, v16bf b, v8f c) {
  // v_wmma_f32_16x16x32_bf16  D = A(16x32) * B(32x16) + C(16x16 f32)
  return __builtin_amdgcn_wmma_f32_16x16x32_bf16(
      /*neg_a=*/false, a, /*neg_b=*/false, b,
      /*c_mod=*/(short)0, c, /*reuse_a=*/false, /*reuse_b=*/false);
}

DEV float gelu_new(float x) {
  float x3 = x * x * x;
  return 0.5f * x * (1.0f + tanhf(0.7978845608028654f * (x + 0.044715f * x3)));
}

// ---------------------------------------------------------------------------
// Hypernetwork weight generation: w = tanh(sub0*wl0 + sub1*wl1)*W + Bp
// ---------------------------------------------------------------------------
__global__ __launch_bounds__(256)
void genw_kernel(const float* __restrict__ sub, const float* __restrict__ wl,
                 const float* __restrict__ Wm,  const float* __restrict__ Bp,
                 bf16_t* __restrict__ out, int n) {
  int i = blockIdx.x * 256 + threadIdx.x;
  if (i >= n) return;
  float p = sub[i] * wl[0] + sub[n + i] * wl[1];
  out[i] = f2bf(tanhf(p) * Wm[i] + Bp[i]);
}

__global__ __launch_bounds__(256)
void genb_kernel(const float* __restrict__ sub, const float* __restrict__ wl,
                 const float* __restrict__ Wm,  const float* __restrict__ Bp,
                 float* __restrict__ out, int n) {
  int i = blockIdx.x * 256 + threadIdx.x;
  if (i >= n) return;
  float p = sub[i] * wl[0] + sub[n + i] * wl[1];
  out[i] = tanhf(p) * Wm[i] + Bp[i];
}

// ---------------------------------------------------------------------------
// LayerNorm over D=1024, one block (8 waves) per row; bf16 output
// ---------------------------------------------------------------------------
__global__ __launch_bounds__(256)
void ln_kernel(const float* __restrict__ x, const float* __restrict__ g,
               const float* __restrict__ bta, bf16_t* __restrict__ y) {
  const int row = blockIdx.x;
  const int tid = threadIdx.x;
  const float* xr = x + (size_t)row * 1024;
  float v[4];
  float s = 0.f, s2 = 0.f;
#pragma unroll
  for (int i = 0; i < 4; ++i) {
    v[i] = xr[tid + 256 * i];
    s += v[i]; s2 += v[i] * v[i];
  }
#pragma unroll
  for (int m = 16; m >= 1; m >>= 1) {
    s  += __shfl_xor(s,  m, 32);
    s2 += __shfl_xor(s2, m, 32);
  }
  __shared__ float rs[8], rs2[8];
  if ((tid & 31) == 0) { rs[tid >> 5] = s; rs2[tid >> 5] = s2; }
  __syncthreads();
  float ts = 0.f, ts2 = 0.f;
#pragma unroll
  for (int w = 0; w < 8; ++w) { ts += rs[w]; ts2 += rs2[w]; }
  float mean = ts * (1.0f / 1024.0f);
  float var  = ts2 * (1.0f / 1024.0f) - mean * mean;
  float rstd = rsqrtf(var + 1e-5f);
  bf16_t* yr = y + (size_t)row * 1024;
#pragma unroll
  for (int i = 0; i < 4; ++i) {
    int c = tid + 256 * i;
    yr[c] = f2bf((v[i] - mean) * rstd * g[c] + bta[c]);
  }
}

// ---------------------------------------------------------------------------
// GEMM  C[M,N] = A[M,K] (bf16) @ W[N,K]^T (bf16) + bias  (+ residual | gelu)
// block tile 128x128, 8 waves, each wave 64x32 (4x2 WMMA accumulators).
// A tile staged to LDS via async global->LDS DMA (pure byte copy).
// ---------------------------------------------------------------------------
constexpr int EPI_NONE = 0, EPI_RES = 1, EPI_GELU = 2;

template<int EPI, bool OUTBF>
__global__ __launch_bounds__(256)
void gemm_bf16_kernel(const bf16_t* __restrict__ A, const bf16_t* __restrict__ W,
                      const float* __restrict__ bias, const float* __restrict__ res,
                      void* __restrict__ Cout, int M, int N, int K) {
  __shared__ bf16_t As[128 * 40];            // 128 rows x 32 k, padded to 40
  const int tid  = threadIdx.x;
  const int lane = tid & 31;
  const int wid  = tid >> 5;
  const int wm   = wid & 1;                  // 2 waves along M
  const int wn   = wid >> 1;                 // 4 waves along N
  const int mblk = blockIdx.y * 128;
  const int nblk = blockIdx.x * 128;
  const int lm   = lane & 15;
  const int hi   = lane >> 4;

  const v8f zero = {0.f,0.f,0.f,0.f,0.f,0.f,0.f,0.f};
  v8f acc[4][2];
#pragma unroll
  for (int i = 0; i < 4; ++i)
#pragma unroll
    for (int j = 0; j < 2; ++j) acc[i][j] = zero;

  const int nK = K >> 5;
  for (int kb = 0; kb < nK; ++kb) {
    __syncthreads();
    // stage A tile: 128 rows x 32 bf16 = 8 KB, 2 x 16B per thread
#pragma unroll
    for (int i = 0; i < 2; ++i) {
      int idx = tid + 256 * i;               // 16B-chunk index 0..511
      int row = idx >> 2;
      int q   = idx & 3;
      copy16_g2l(&As[row * 40 + q * 8],
                 A + (size_t)(mblk + row) * K + kb * 32 + q * 8);
    }
    wait_async0();
    __syncthreads();

    // B fragments straight from global bf16 (lane = column N, K 8h..8h+7 and 16+8h..16+8h+7)
    v16bf bfrag[2];
#pragma unroll
    for (int nt = 0; nt < 2; ++nt) {
      int n = nblk + wn * 32 + nt * 16 + lm;
      const bf16_t* src = W + (size_t)n * K + kb * 32 + hi * 8;
      union { v16bf v; v8bf h[2]; } u;
      u.h[0] = *(const v8bf*)(src);
      u.h[1] = *(const v8bf*)(src + 16);
      bfrag[nt] = u.v;
    }
    // A fragments from LDS (ISA 16-bit A 16x32 swizzle)
    v16bf afrag[4];
#pragma unroll
    for (int mt = 0; mt < 4; ++mt) {
      const bf16_t* srow = &As[(wm * 64 + mt * 16 + lm) * 40];
      union { v16bf v; unsigned u[8]; } ua;
#pragma unroll
      for (int vv = 0; vv < 8; ++vv) {
        int kp = (vv < 4 ? 2 * vv : 16 + 2 * (vv - 4)) + hi * 8;
        ua.u[vv] = *(const unsigned*)(srow + kp);
      }
      afrag[mt] = ua.v;
    }
#pragma unroll
    for (int mt = 0; mt < 4; ++mt)
#pragma unroll
      for (int nt = 0; nt < 2; ++nt)
        acc[mt][nt] = wmma_bf16(afrag[mt], bfrag[nt], acc[mt][nt]);
  }

  // epilogue: C-fragment layout lane->N, vgpr r -> M = r + 8*hi
#pragma unroll
  for (int mt = 0; mt < 4; ++mt) {
#pragma unroll
    for (int nt = 0; nt < 2; ++nt) {
      int gcol = nblk + wn * 32 + nt * 16 + lm;
      float bv = bias[gcol];
#pragma unroll
      for (int r = 0; r < 8; ++r) {
        int grow = mblk + wm * 64 + mt * 16 + r + 8 * hi;
        float val = acc[mt][nt][r] + bv;
        if (EPI == EPI_GELU) val = gelu_new(val);
        if (EPI == EPI_RES)  val += res[(size_t)grow * N + gcol];
        if (OUTBF) ((bf16_t*)Cout)[(size_t)grow * N + gcol] = f2bf(val);
        else       ((float*)Cout)[(size_t)grow * N + gcol]  = val;
      }
    }
  }
}

// ---------------------------------------------------------------------------
// Flash attention: grid (16 q-blocks, 32 bh). Block = 8 waves, wave = 16 q rows.
// qkv (bf16) layout: [b, s, 3*1024], q at h*64, k at 1024+h*64, v at 2048+h*64.
// Output bf16 [b*2048+s, 1024].
// ---------------------------------------------------------------------------
__global__ __launch_bounds__(256)
void attn_kernel(const bf16_t* __restrict__ qkv, bf16_t* __restrict__ out) {
  __shared__ bf16_t k_lds [32 * 72];         // [key][hd], padded (rows 144B)
  __shared__ bf16_t vT_lds[64 * 40];         // [hd][key], padded (rows 80B)
  __shared__ bf16_t p_lds [8 * 16 * 36];     // per wave [row][key], padded

  const int tid  = threadIdx.x, lane = tid & 31, wid = tid >> 5;
  const int lm   = lane & 15,   hi   = lane >> 4;
  const int bh   = blockIdx.y;
  const int b    = bh >> 4, h = bh & 15;
  const int qb0  = blockIdx.x * 128 + wid * 16;
  const float scale = 0.125f;                // 1/sqrt(64)

  // preload q A-fragments for hd halves {0..31, 32..63} (packed pair loads)
  v16bf aq[2];
  {
    const bf16_t* qbase = qkv + (size_t)(b * 2048 + qb0 + lm) * 3072 + h * 64;
#pragma unroll
    for (int half = 0; half < 2; ++half) {
      union { v16bf v; unsigned u[8]; } ua;
#pragma unroll
      for (int vv = 0; vv < 8; ++vv) {
        int kp = (vv < 4 ? 2 * vv : 16 + 2 * (vv - 4)) + hi * 8 + half * 32;
        ua.u[vv] = *(const unsigned*)(qbase + kp);
      }
      aq[half] = ua.v;
    }
  }

  const v8f zero = {0.f,0.f,0.f,0.f,0.f,0.f,0.f,0.f};
  v8f co[4]; co[0] = zero; co[1] = zero; co[2] = zero; co[3] = zero;
  float mrow[8], lrow[8];
#pragma unroll
  for (int r = 0; r < 8; ++r) { mrow[r] = -INFINITY; lrow[r] = 0.f; }

  const int nkb = (blockIdx.x + 1) * 4;      // causal: covers block's last row
  for (int kb = 0; kb < nkb; ++kb) {
    __syncthreads();
    // cooperative stage: 32 keys x 64 hd of K (async copy) and V^T (transpose)
    {
      int key = tid >> 3, c = tid & 7;       // 8 x 16B chunks per row
      size_t gr = (size_t)(b * 2048 + kb * 32 + key) * 3072 + h * 64 + c * 8;
      copy16_g2l(&k_lds[key * 72 + c * 8], qkv + gr + 1024);
      union { uint4 q4; bf16_t e[8]; } uv;
      uv.q4 = *(const uint4*)(qkv + gr + 2048);
#pragma unroll
      for (int j = 0; j < 8; ++j)
        vT_lds[(c * 8 + j) * 40 + key] = uv.e[j];
    }
    wait_async0();
    __syncthreads();

    if (kb * 32 > qb0 + 15) continue;        // fully masked for this wave

    // scores S = Q K^T  (2 key sub-tiles x 2 hd k-steps)
    v8f cs[2]; cs[0] = zero; cs[1] = zero;
#pragma unroll
    for (int half = 0; half < 2; ++half) {
#pragma unroll
      for (int nt = 0; nt < 2; ++nt) {
        const bf16_t* src = &k_lds[(nt * 16 + lm) * 72 + half * 32 + hi * 8];
        union { v16bf v; v8bf hh[2]; } u;
        u.hh[0] = *(const v8bf*)(src);
        u.hh[1] = *(const v8bf*)(src + 16);
        cs[nt] = wmma_bf16(aq[half], u.v, cs[nt]);
      }
    }

    // scale + causal mask + online softmax (xor-shuffles stay in 16-lane half)
    float sc[2][8];
#pragma unroll
    for (int r = 0; r < 8; ++r) {
      int qrow = qb0 + r + 8 * hi;
      float rmax = -INFINITY;
#pragma unroll
      for (int nt = 0; nt < 2; ++nt) {
        int key = kb * 32 + nt * 16 + lm;
        float s = cs[nt][r] * scale;
        if (key > qrow) s = -10000.0f;
        sc[nt][r] = s;
        rmax = fmaxf(rmax, s);
      }
#pragma unroll
      for (int msk = 8; msk >= 1; msk >>= 1)
        rmax = fmaxf(rmax, __shfl_xor(rmax, msk, 32));

      float mnew  = fmaxf(mrow[r], rmax);
      float alpha = __expf(mrow[r] - mnew);
      mrow[r] = mnew;
      float psum = 0.f;
#pragma unroll
      for (int nt = 0; nt < 2; ++nt) {
        float p = __expf(sc[nt][r] - mnew);
        sc[nt][r] = p;
        psum += p;
      }
#pragma unroll
      for (int msk = 8; msk >= 1; msk >>= 1)
        psum += __shfl_xor(psum, msk, 32);
      lrow[r] = lrow[r] * alpha + psum;
#pragma unroll
      for (int j = 0; j < 4; ++j) co[j][r] *= alpha;
    }

    // C-fragment -> A-fragment relayout of P via per-wave LDS
    bf16_t* pw = &p_lds[wid * 16 * 36];
#pragma unroll
    for (int nt = 0; nt < 2; ++nt)
#pragma unroll
      for (int r = 0; r < 8; ++r)
        pw[(r + 8 * hi) * 36 + nt * 16 + lm] = f2bf(sc[nt][r]);
    asm volatile("s_wait_dscnt 0" ::: "memory");   // in-wave DS ordering

    v16bf ap;
    {
      const bf16_t* pr = &p_lds[wid * 16 * 36 + lm * 36];
      union { v16bf v; unsigned u[8]; } ua;
#pragma unroll
      for (int vv = 0; vv < 8; ++vv) {
        int kp = (vv < 4 ? 2 * vv : 16 + 2 * (vv - 4)) + hi * 8;
        ua.u[vv] = *(const unsigned*)(pr + kp);
      }
      ap = ua.v;
    }

    // O += P @ V  (4 hd sub-tiles)
#pragma unroll
    for (int j = 0; j < 4; ++j) {
      const bf16_t* src = &vT_lds[(j * 16 + lm) * 40 + hi * 8];
      union { v16bf v; v8bf hh[2]; } u;
      u.hh[0] = *(const v8bf*)(src);
      u.hh[1] = *(const v8bf*)(src + 16);
      co[j] = wmma_bf16(ap, u.v, co[j]);
    }
  }

  // normalize and store bf16: out[b*2048+q, h*64+hd]
#pragma unroll
  for (int j = 0; j < 4; ++j) {
    int col = h * 64 + j * 16 + lm;
#pragma unroll
    for (int r = 0; r < 8; ++r) {
      int grow = b * 2048 + qb0 + r + 8 * hi;
      out[(size_t)grow * 1024 + col] = f2bf(co[j][r] / lrow[r]);
    }
  }
}

// ---------------------------------------------------------------------------
// Host orchestration
// ---------------------------------------------------------------------------
extern "C" void kernel_launch(void* const* d_in, const int* in_sizes, int n_in,
                              void* d_out, int out_size, void* d_ws, size_t ws_size,
                              hipStream_t stream) {
  (void)in_sizes; (void)n_in; (void)out_size; (void)ws_size;
  const float* hs   = (const float*)d_in[0];
  const float* ln1g = (const float*)d_in[33];
  const float* ln1b = (const float*)d_in[34];
  const float* ln2g = (const float*)d_in[35];
  const float* ln2b = (const float*)d_in[36];

  char* ws = (char*)d_ws;
  bf16_t* w_ca = (bf16_t*)(ws + 0);            // 3072x1024 bf16  (6,291,456 B)
  bf16_t* w_cp = (bf16_t*)(ws + 6291456);      // 1024x1024 bf16  (2,097,152 B)
  bf16_t* w_cf = (bf16_t*)(ws + 8388608);      // 4096x1024 bf16  (8,388,608 B)
  bf16_t* w_cm = (bf16_t*)(ws + 16777216);     // 1024x4096 bf16  (8,388,608 B)
  float*  b_ca = (float*)(ws + 25165824);      // 3072 f32
  float*  b_cp = (float*)(ws + 25178112);      // 1024 f32
  float*  b_cf = (float*)(ws + 25182208);      // 4096 f32
  float*  b_cm = (float*)(ws + 25198592);      // 1024 f32
  char* act = ws + 25202688;
  bf16_t* h1    = (bf16_t*)(act);              // 4096x1024 bf16 ( 8,388,608 B)
  bf16_t* qkvb  = (bf16_t*)(act + 8388608);    // 4096x3072 bf16 (25,165,824 B)
  bf16_t* mact  = (bf16_t*)(act);              // 4096x4096 bf16, aliases h1+qkv
  bf16_t* attno = (bf16_t*)(act + 33554432);   // 4096x1024 bf16 ( 8,388,608 B)
  bf16_t* h2    = attno;                       // aliases attno (dead by LN2)
  float*  outf  = (float*)d_out;

  // hypernetwork weight generation (bf16 weights, f32 biases)
  auto gen = [&](int base, bf16_t* wout, float* bout, int nw, int nb) {
    genw_kernel<<<(nw + 255) / 256, 256, 0, stream>>>(
        (const float*)d_in[base + 0], (const float*)d_in[base + 1],
        (const float*)d_in[base + 2], (const float*)d_in[base + 3], wout, nw);
    genb_kernel<<<(nb + 255) / 256, 256, 0, stream>>>(
        (const float*)d_in[base + 4], (const float*)d_in[base + 5],
        (const float*)d_in[base + 6], (const float*)d_in[base + 7], bout, nb);
  };
  gen(1,  w_ca, b_ca, 3072 * 1024, 3072);
  gen(9,  w_cp, b_cp, 1024 * 1024, 1024);
  gen(17, w_cf, b_cf, 4096 * 1024, 4096);
  gen(25, w_cm, b_cm, 1024 * 4096, 1024);

  // LN1 -> bf16
  ln_kernel<<<4096, 256, 0, stream>>>(hs, ln1g, ln1b, h1);
  // qkv = h1 @ w_ca^T + b_ca   (bf16 out)
  gemm_bf16_kernel<EPI_NONE, true><<<dim3(24, 32), 256, 0, stream>>>(
      h1, w_ca, b_ca, nullptr, qkvb, 4096, 3072, 1024);
  // flash attention (bf16 in/out)
  attn_kernel<<<dim3(16, 32), 256, 0, stream>>>(qkvb, attno);
  // hs2 = attno @ w_cp^T + b_cp + hs   -> d_out (f32 residual stream)
  gemm_bf16_kernel<EPI_RES, false><<<dim3(8, 32), 256, 0, stream>>>(
      attno, w_cp, b_cp, hs, outf, 4096, 1024, 1024);
  // LN2 -> bf16
  ln_kernel<<<4096, 256, 0, stream>>>(outf, ln2g, ln2b, h2);
  // m = gelu(h2 @ w_cf^T + b_cf)   (bf16 out)
  gemm_bf16_kernel<EPI_GELU, true><<<dim3(32, 32), 256, 0, stream>>>(
      h2, w_cf, b_cf, nullptr, mact, 4096, 4096, 1024);
  // out = m @ w_cm^T + b_cm + hs2  (in-place residual in d_out)
  gemm_bf16_kernel<EPI_RES, false><<<dim3(8, 32), 256, 0, stream>>>(
      mact, w_cm, b_cm, outf, outf, 4096, 1024, 4096);
}
